// GLAMemoryReader_53687091200017
// MI455X (gfx1250) — matrix operand order repaired and checked by reading, not compiled, not run on hardware
//
#include <hip/hip_runtime.h>

typedef __bf16 bf16_t;
typedef __attribute__((ext_vector_type(16))) __bf16 v16bf;
typedef __attribute__((ext_vector_type(8)))  __bf16 v8bf;
typedef __attribute__((ext_vector_type(8)))  float  v8f;

#define EPSV 1e-5f

constexpr int Bn  = 4;
constexpr int Tn  = 8192;
constexpr int Hn  = 1024;
constexpr int KDn = 512;
constexpr int VDn = 1024;
constexpr int NT  = Bn * Tn;   // 32768 tokens
constexpr int KSTEPS = Hn / 32;

#define WMMA_BF16(a, b, c) \
  __builtin_amdgcn_wmma_f32_16x16x32_bf16(false, (a), false, (b), (short)0, (c), false, false)

// Async global->LDS copies. Passing the real LDS pointer (ptrtoint) both yields the
// correct wave-relative LDS byte offset (generic-pointer low 32 bits, ISA 10.2) and
// escapes the __shared__ object so the "memory" clobber keeps later ds_loads alive.
__device__ __forceinline__ void async_to_lds_b128(const void* gptr, void* lptr) {
  unsigned loff = (unsigned)(size_t)lptr;
  asm volatile("global_load_async_to_lds_b128 %0, %1, off"
               :: "v"(loff), "v"(gptr) : "memory");
}
__device__ __forceinline__ void async_to_lds_b64(const void* gptr, void* lptr) {
  unsigned loff = (unsigned)(size_t)lptr;
  asm volatile("global_load_async_to_lds_b64 %0, %1, off"
               :: "v"(loff), "v"(gptr) : "memory");
}
__device__ __forceinline__ void wait_async0() {
  asm volatile("s_wait_asynccnt 0x0" ::: "memory");
}

// ---------------- f32 -> bf16 conversion ----------------
__global__ void k_cvt_bf16(const float* __restrict__ in, bf16_t* __restrict__ out, int n) {
  int i = blockIdx.x * blockDim.x + threadIdx.x;
  if (i < n) out[i] = (bf16_t)in[i];
}

// ---------------- RMSNorm over H=1024, one block (256 thr) per token ----------------
__global__ void k_rmsnorm_x(const float* __restrict__ x, const float* __restrict__ w,
                            bf16_t* __restrict__ xbf) {
  __shared__ float red[256];
  int t = blockIdx.x;
  const float4 v  = ((const float4*)(x + (size_t)t * Hn))[threadIdx.x];
  float ss = v.x*v.x + v.y*v.y + v.z*v.z + v.w*v.w;
  red[threadIdx.x] = ss;
  __syncthreads();
  for (int s = 128; s > 0; s >>= 1) {
    if (threadIdx.x < s) red[threadIdx.x] += red[threadIdx.x + s];
    __syncthreads();
  }
  float scale = rsqrtf(red[0] * (1.0f / Hn) + EPSV);
  const float4 wv = ((const float4*)w)[threadIdx.x];
  bf16_t* op = xbf + (size_t)t * Hn + threadIdx.x * 4;
  op[0] = (bf16_t)(v.x * scale * wv.x);
  op[1] = (bf16_t)(v.y * scale * wv.y);
  op[2] = (bf16_t)(v.z * scale * wv.z);
  op[3] = (bf16_t)(v.w * scale * wv.w);
}

// ---------------- WMMA fragment loaders (contiguous k = 16h + s order, wave32) ----------------
__device__ __forceinline__ v16bf load_a16x32(const bf16_t* __restrict__ A, int lda,
                                             int row0, int k0, int lane) {
  int m = lane & 15, h = lane >> 4;
  const bf16_t* p = A + (size_t)(row0 + m) * lda + (k0 + h * 16);
  v8bf c0 = *(const v8bf*)p;        // K = base .. base+7   -> frag[0..7]
  v8bf c1 = *(const v8bf*)(p + 8);  // K = base+8..base+15  -> frag[8..15]
  v16bf f;
#pragma unroll
  for (int i = 0; i < 8; ++i) { f[i] = c0[i]; f[i + 8] = c1[i]; }
  return f;
}

// ---------------- q projection: Q[NT,512] = Xbf @ Wq^T ----------------
// 8 waves along M (32 rows each); whole block shares one 32-col B panel staged
// through LDS with double-buffered async copies.
__global__ void k_qgemm(const bf16_t* __restrict__ X, const bf16_t* __restrict__ Wq,
                        float* __restrict__ Q) {
  __shared__ bf16_t sB[2][32][32];           // [buf][col][k] = 4 KB
  int tid = threadIdx.x, lane = tid & 31, wave = tid >> 5;
  int row0 = (blockIdx.y * 8 + wave) * 32;
  int col0 = blockIdx.x * 32;
  int scol = tid >> 3, skc = tid & 7;        // each thread stages one 8B chunk
  const bf16_t* gq = Wq + (size_t)(col0 + scol) * Hn + skc * 4;

  v8f a00 = {}, a01 = {}, a10 = {}, a11 = {};

  async_to_lds_b64(gq, &sB[0][scol][skc * 4]);
  wait_async0();
  __syncthreads();

  for (int ks = 0; ks < KSTEPS; ++ks) {
    int k0 = ks * 32, cur = ks & 1;
    if (ks + 1 < KSTEPS)
      async_to_lds_b64(gq + (k0 + 32), &sB[cur ^ 1][scol][skc * 4]);

    v16bf af0 = load_a16x32(X, Hn, row0,      k0, lane);
    v16bf af1 = load_a16x32(X, Hn, row0 + 16, k0, lane);
    int bc = lane & 15, kb = (lane >> 4) * 16;
    v16bf bf0 = *(const v16bf*)&sB[cur][bc][kb];
    v16bf bf1 = *(const v16bf*)&sB[cur][bc + 16][kb];

    a00 = WMMA_BF16(af0, bf0, a00);
    a01 = WMMA_BF16(af0, bf1, a01);
    a10 = WMMA_BF16(af1, bf0, a10);
    a11 = WMMA_BF16(af1, bf1, a11);

    wait_async0();
    __syncthreads();
  }

  int n = lane & 15, rb = 8 * (lane >> 4);
#pragma unroll
  for (int v = 0; v < 8; ++v) {
    Q[(size_t)(row0 +      v + rb) * KDn + col0 +      n] = a00[v];
    Q[(size_t)(row0 +      v + rb) * KDn + col0 + 16 + n] = a01[v];
    Q[(size_t)(row0 + 16 + v + rb) * KDn + col0 +      n] = a10[v];
    Q[(size_t)(row0 + 16 + v + rb) * KDn + col0 + 16 + n] = a11[v];
  }
}

// ---------------- per-token head mixing (normalize, state einsum, head RMSNorm) ----------------
__global__ void k_heads(const float* __restrict__ Q, const float* __restrict__ S,
                        const float* __restrict__ vw, bf16_t* __restrict__ O) {
  __shared__ float sq[512];
  __shared__ float sscale[16];
  __shared__ float so[1024];
  int t = blockIdx.x;
  int b = t >> 13;
  int tid = threadIdx.x;
  const float* qp = Q + (size_t)t * KDn;
  sq[2 * tid]     = qp[2 * tid];
  sq[2 * tid + 1] = qp[2 * tid + 1];
  __syncthreads();
  if (tid < 16) {
    float ss = 0.f;
#pragma unroll
    for (int k = 0; k < 32; ++k) { float q = sq[tid * 32 + k]; ss += q * q; }
    sscale[tid] = 1.0f / fmaxf(sqrtf(ss), 1e-12f);
  }
  __syncthreads();
  int h = tid >> 4, vg = tid & 15;
  const float* sp = S + (size_t)(b * 16 + h) * 32 * 64;
  float ninv = sscale[h];
  float a0 = 0.f, a1 = 0.f, a2 = 0.f, a3 = 0.f;
#pragma unroll 4
  for (int k = 0; k < 32; ++k) {
    float qn = sq[h * 32 + k] * ninv;
    const float* sr = sp + k * 64 + vg;
    a0 += qn * sr[0]; a1 += qn * sr[16]; a2 += qn * sr[32]; a3 += qn * sr[48];
  }
  so[h * 64 + vg]      = a0;
  so[h * 64 + vg + 16] = a1;
  so[h * 64 + vg + 32] = a2;
  so[h * 64 + vg + 48] = a3;
  __syncthreads();
  if (tid < 16) {
    float ss = 0.f;
#pragma unroll
    for (int k = 0; k < 64; ++k) { float o = so[tid * 64 + k]; ss += o * o; }
    sscale[tid] = rsqrtf(ss * (1.0f / 64.0f) + EPSV);
  }
  __syncthreads();
  float sc = sscale[h];
  bf16_t* op = O + (size_t)t * VDn + h * 64 + vg;
  op[0]  = (bf16_t)(a0 * sc * vw[vg]);
  op[16] = (bf16_t)(a1 * sc * vw[vg + 16]);
  op[32] = (bf16_t)(a2 * sc * vw[vg + 32]);
  op[48] = (bf16_t)(a3 * sc * vw[vg + 48]);
}

// ------- fused gate+output GEMMs, double-buffered async-LDS B staging -------
__device__ __forceinline__ void store_gated(float* __restrict__ out, int row0, int col0,
                                            int lane, const v8f& g, const v8f& o) {
  int n = col0 + (lane & 15), rb = 8 * (lane >> 4);
#pragma unroll
  for (int v = 0; v < 8; ++v) {
    float s = 1.0f / (1.0f + __expf(-g[v]));
    out[(size_t)(row0 + v + rb) * VDn + n] = s * o[v];
  }
}

__global__ void k_outgemm(const bf16_t* __restrict__ X, const bf16_t* __restrict__ O,
                          const bf16_t* __restrict__ Wg, const bf16_t* __restrict__ Wo,
                          float* __restrict__ out) {
  __shared__ bf16_t sB[2][2][64][32];        // [buf][mat][col][k] = 16 KB
  int tid  = threadIdx.x;
  int lane = tid & 31, wave = tid >> 5;
  int wm = wave & 3, wn = wave >> 2;         // 4 waves along M, 2 along N
  int row0  = blockIdx.y * 128 + wm * 32;
  int col0b = blockIdx.x * 64;
  int col0  = col0b + wn * 32;

  int scol = tid >> 2, skc = tid & 3;        // 16B chunk per thread per matrix
  const bf16_t* gq = Wg + (size_t)(col0b + scol) * Hn + skc * 8;
  const bf16_t* go = Wo + (size_t)(col0b + scol) * Hn + skc * 8;

  v8f g00 = {}, g01 = {}, g10 = {}, g11 = {};
  v8f o00 = {}, o01 = {}, o10 = {}, o11 = {};

  async_to_lds_b128(gq, &sB[0][0][scol][skc * 8]);
  async_to_lds_b128(go, &sB[0][1][scol][skc * 8]);
  wait_async0();
  __syncthreads();

  for (int ks = 0; ks < KSTEPS; ++ks) {
    int k0 = ks * 32, cur = ks & 1;
    if (ks + 1 < KSTEPS) {                   // prefetch next panel while computing
      async_to_lds_b128(gq + (k0 + 32), &sB[cur ^ 1][0][scol][skc * 8]);
      async_to_lds_b128(go + (k0 + 32), &sB[cur ^ 1][1][scol][skc * 8]);
    }

    v16bf ax0 = load_a16x32(X, Hn, row0,      k0, lane);
    v16bf ax1 = load_a16x32(X, Hn, row0 + 16, k0, lane);
    v16bf ao0 = load_a16x32(O, Hn, row0,      k0, lane);
    v16bf ao1 = load_a16x32(O, Hn, row0 + 16, k0, lane);

    int bc = wn * 32 + (lane & 15), kb = (lane >> 4) * 16;
    v16bf bg0 = *(const v16bf*)&sB[cur][0][bc][kb];
    v16bf bg1 = *(const v16bf*)&sB[cur][0][bc + 16][kb];
    v16bf bo0 = *(const v16bf*)&sB[cur][1][bc][kb];
    v16bf bo1 = *(const v16bf*)&sB[cur][1][bc + 16][kb];

    g00 = WMMA_BF16(ax0, bg0, g00);
    g01 = WMMA_BF16(ax0, bg1, g01);
    g10 = WMMA_BF16(ax1, bg0, g10);
    g11 = WMMA_BF16(ax1, bg1, g11);
    o00 = WMMA_BF16(ao0, bo0, o00);
    o01 = WMMA_BF16(ao0, bo1, o01);
    o10 = WMMA_BF16(ao1, bo0, o10);
    o11 = WMMA_BF16(ao1, bo1, o11);

    wait_async0();
    __syncthreads();
  }

  store_gated(out, row0,      col0,      lane, g00, o00);
  store_gated(out, row0,      col0 + 16, lane, g01, o01);
  store_gated(out, row0 + 16, col0,      lane, g10, o10);
  store_gated(out, row0 + 16, col0 + 16, lane, g11, o11);
}

extern "C" void kernel_launch(void* const* d_in, const int* in_sizes, int n_in,
                              void* d_out, int out_size, void* d_ws, size_t ws_size,
                              hipStream_t stream) {
  const float* hidden = (const float*)d_in[0];   // [4,8192,1024]
  const float* state  = (const float*)d_in[1];   // [4,16,32,64]
  const float* norm_w = (const float*)d_in[2];   // [1024]
  const float* q_w    = (const float*)d_in[3];   // [512,1024]
  const float* v_norm = (const float*)d_in[4];   // [64]
  const float* o_w    = (const float*)d_in[5];   // [1024,1024]
  const float* gate_w = (const float*)d_in[6];   // [1024,1024]
  float* out = (float*)d_out;

  char* ws = (char*)d_ws;
  bf16_t* xbf  = (bf16_t*)(ws);                          // 64 MB
  float*  qbuf = (float*) (ws + ((size_t)64  << 20));    // 64 MB
  bf16_t* obf  = (bf16_t*)(ws + ((size_t)128 << 20));    // 64 MB
  bf16_t* qwbf = (bf16_t*)(ws + ((size_t)192 << 20));    //  1 MB
  bf16_t* gwbf = (bf16_t*)(ws + ((size_t)193 << 20));    //  2 MB
  bf16_t* owbf = (bf16_t*)(ws + ((size_t)195 << 20));    //  2 MB

  k_cvt_bf16<<<(KDn * Hn + 255) / 256, 256, 0, stream>>>(q_w,    qwbf, KDn * Hn);
  k_cvt_bf16<<<(Hn  * Hn + 255) / 256, 256, 0, stream>>>(gate_w, gwbf, Hn * Hn);
  k_cvt_bf16<<<(Hn  * VDn + 255) / 256, 256, 0, stream>>>(o_w,   owbf, Hn * VDn);

  k_rmsnorm_x<<<NT, 256, 0, stream>>>(hidden, norm_w, xbf);

  k_qgemm<<<dim3(KDn / 32, NT / 256), 256, 0, stream>>>(xbf, qwbf, qbuf);

  k_heads<<<NT, 256, 0, stream>>>(qbuf, state, v_norm, obf);

  k_outgemm<<<dim3(Hn / 64, NT / 128), 256, 0, stream>>>(xbf, obf, gwbf, owbf, out);
}
